// MultiGRU_24515673325789
// MI455X (gfx1250) — matrix operand order repaired and checked by reading, not compiled, not run on hardware
//
#include <hip/hip_runtime.h>
#include <hip/hip_bf16.h>
#include <math.h>

typedef __bf16 bf16_t;
typedef __attribute__((ext_vector_type(16))) __bf16 v16bf;
typedef __attribute__((ext_vector_type(8)))  float  v8f;
typedef __attribute__((ext_vector_type(4)))  unsigned int v4u;

#define B_   256
#define T_   112
#define H_   1024
#define IN_  64
#define OUT_ 64
#define G3_  3072
#define BH_  (B_*H_)
#define LAG_ 7

// ---------------- workspace layout (bytes) ----------------
constexpr size_t SZ_WIH0B = (size_t)G3_*IN_*2;   // w_ih1 bf16
constexpr size_t SZ_WB    = (size_t)G3_*H_*2;    // 3072x1024 bf16
constexpr size_t SZ_XB    = (size_t)B_*T_*IN_*2; // x bf16
constexpr size_t SZ_HF    = (size_t)BH_*4;       // hidden fp32
constexpr size_t SZ_HB    = (size_t)BH_*2;       // hidden bf16 mirror
constexpr size_t SZ_GF    = (size_t)B_*G3_*4;    // gate scratch fp32

constexpr size_t OFF_WIHB0 = 0;
constexpr size_t OFF_WIHB1 = OFF_WIHB0 + SZ_WIH0B;
constexpr size_t OFF_WIHB2 = OFF_WIHB1 + SZ_WB;
constexpr size_t OFF_WIHB3 = OFF_WIHB2 + SZ_WB;
constexpr size_t OFF_WHHB0 = OFF_WIHB3 + SZ_WB;
constexpr size_t OFF_WHHB1 = OFF_WHHB0 + SZ_WB;
constexpr size_t OFF_WHHB2 = OFF_WHHB1 + SZ_WB;
constexpr size_t OFF_WHHB3 = OFF_WHHB2 + SZ_WB;
constexpr size_t OFF_XBF   = OFF_WHHB3 + SZ_WB;
constexpr size_t OFF_H0    = OFF_XBF + SZ_XB;     // h1..h4 contiguous fp32
constexpr size_t OFF_HBF0  = OFF_H0 + 4*SZ_HF;    // bf16 mirrors contiguous
constexpr size_t OFF_GI0   = OFF_HBF0 + 4*SZ_HB;  // gi[4] contiguous
constexpr size_t OFF_GH0   = OFF_GI0 + 4*SZ_GF;   // gh[4] contiguous
constexpr size_t OFF_RING  = OFF_GH0 + 4*SZ_GF;   // 7 x BH fp32 (h2 lag buffer)
constexpr size_t OFF_AUX   = OFF_RING + (size_t)LAG_*SZ_HF; // 3 floats
// total ~83 MB

// ---------------- WMMA fragment helpers ----------------
union FragU { v16bf v; v4u u[2]; };

__device__ static inline v16bf load_a_frag(const bf16_t* __restrict__ A, int lda,
                                           int m0, int k0, int lane) {
  // 16x32 bf16 A tile: lanes 0-15 -> K {0..7,16..23}; lanes 16-31 -> K {8..15,24..31}
  const int r    = lane & 15;
  const int half = lane >> 4;
  const bf16_t* p = A + (size_t)(m0 + r) * lda + (k0 + half * 8);
  FragU f;
  f.u[0] = *(const v4u*)p;          // K = k0+half*8 .. +7
  f.u[1] = *(const v4u*)(p + 16);   // K = k0+half*8+16 .. +23
  return f.v;
}

__device__ static inline v16bf load_b_frag(const bf16_t* __restrict__ W, int ldk,
                                           int n0, int k0, int lane) {
  // 32x16 bf16 B tile from weight stored [N,K] row-major (B = W^T):
  // lanes 0-15 -> K {0..7,16..23} of col n0+lane; lanes 16-31 -> K {8..15,24..31}
  const int n    = n0 + (lane & 15);
  const int koff = (lane >> 4) * 8;
  const bf16_t* p = W + (size_t)n * ldk + (k0 + koff);
  FragU f;
  f.u[0] = *(const v4u*)p;          // 8 bf16
  f.u[1] = *(const v4u*)(p + 16);   // 8 bf16 at +16
  return f.v;
}

__device__ static inline void store_c_tile(float* __restrict__ C, int ldc,
                                           int m0, int n0, int lane, const v8f& acc) {
  const int cn = n0 + (lane & 15);
  const int rb = m0 + ((lane >> 4) ? 8 : 0);
#pragma unroll
  for (int v = 0; v < 8; ++v)
    C[(size_t)(rb + v) * ldc + cn] = acc[v];
}

// ---------------- GEMM: C[M,N] (fp32) = A[M,K] (bf16) * W[N,K]^T (bf16) ----------------
struct GemmDesc { const bf16_t* A; const bf16_t* Bm; float* C; int lda; int K; };
struct GemmArgs { GemmDesc d[6]; };

__global__ void __launch_bounds__(256)
wmma_gemm_kernel(GemmArgs args, int N) {
  const GemmDesc g = args.d[blockIdx.z];
  const int lane = threadIdx.x & 31;
  const int wave = threadIdx.x >> 5;
  const int wm = wave >> 2, wn = wave & 3;          // waves: 2(M) x 4(N)
  const int m0 = blockIdx.x * 64 + wm * 32;         // 32 rows / wave
  const int n0 = blockIdx.y * 256 + wn * 64;        // 64 cols / wave

  const v8f vzero = {0.f,0.f,0.f,0.f,0.f,0.f,0.f,0.f};
  v8f acc[2][4];
#pragma unroll
  for (int i = 0; i < 2; ++i)
#pragma unroll
    for (int j = 0; j < 4; ++j) acc[i][j] = vzero;

  const int K = g.K;
  for (int k0 = 0; k0 < K; k0 += 32) {
    v16bf a0 = load_a_frag(g.A, g.lda, m0,      k0, lane);
    v16bf a1 = load_a_frag(g.A, g.lda, m0 + 16, k0, lane);
#pragma unroll
    for (int j = 0; j < 4; ++j) {
      v16bf b = load_b_frag(g.Bm, K, n0 + j * 16, k0, lane);
      acc[0][j] = __builtin_amdgcn_wmma_f32_16x16x32_bf16(false, a0, false, b,
                    (short)0, acc[0][j], false, false);
      acc[1][j] = __builtin_amdgcn_wmma_f32_16x16x32_bf16(false, a1, false, b,
                    (short)0, acc[1][j], false, false);
    }
  }
#pragma unroll
  for (int j = 0; j < 4; ++j) {
    store_c_tile(g.C, N, m0,      n0 + j * 16, lane, acc[0][j]);
    store_c_tile(g.C, N, m0 + 16, n0 + j * 16, lane, acc[1][j]);
  }
}

// ---------------- GRU gate fusion + aux1/aux2 accumulation ----------------
// mode 0: plain (layers 1,4); mode 1: layer2 (lag-7 ring -> aux[0]); mode 2: layer3 (lag-1 -> aux[1])
struct GateDesc { const float* gi; const float* gh; const float* bih; const float* bhh;
                  float* h; bf16_t* hbf; int mode; };
struct GateArgs { GateDesc d[3]; };

__global__ void __launch_bounds__(256)
gru_gate_kernel(GateArgs args, float* ring, float* aux, int t) {
  const GateDesc g = args.d[blockIdx.z];
  const int idx = blockIdx.x * 256 + threadIdx.x;   // b*H + j
  const int b = idx >> 10;
  const int j = idx & (H_ - 1);
  const size_t base = (size_t)b * G3_;

  float ir  = g.gi[base + j]          + g.bih[j];
  float iz  = g.gi[base + H_ + j]     + g.bih[H_ + j];
  float inn = g.gi[base + 2*H_ + j]   + g.bih[2*H_ + j];
  float hr  = g.gh[base + j]          + g.bhh[j];
  float hz  = g.gh[base + H_ + j]     + g.bhh[H_ + j];
  float hn  = g.gh[base + 2*H_ + j]   + g.bhh[2*H_ + j];

  float r = 1.f / (1.f + expf(-(ir + hr)));
  float z = 1.f / (1.f + expf(-(iz + hz)));
  float n = tanhf(inn + r * hn);
  float hold = g.h[idx];
  float hnew = (1.f - z) * n + z * hold;
  g.h[idx]   = hnew;
  g.hbf[idx] = (bf16_t)hnew;

  float d2 = 0.f;
  bool contrib = false;
  if (g.mode == 1) {
    float* slot = ring + (size_t)(t % LAG_) * BH_ + idx;
    if (t >= LAG_) { float d = hnew - *slot; d2 = d * d; contrib = true; }
    *slot = hnew;
  } else if (g.mode == 2) {
    if (t >= 1) { float d = hnew - hold; d2 = d * d; contrib = true; }
  }
  if (g.mode != 0) {
    __shared__ float sred[256];
    sred[threadIdx.x] = d2;
    __syncthreads();
    for (int s = 128; s > 0; s >>= 1) {
      if (threadIdx.x < s) sred[threadIdx.x] += sred[threadIdx.x + s];
      __syncthreads();
    }
    if (threadIdx.x == 0 && contrib)
      atomicAdd(aux + (g.mode == 1 ? 0 : 1), sred[0] * (1.f / (float)BH_));
  }
}

// ---------------- output heads + aux3 (tiny: ~1% of FLOPs, VALU) ----------------
__global__ void __launch_bounds__(256)
heads_kernel(const float* __restrict__ h2, const float* __restrict__ h3,
             const float* __restrict__ h4,
             const float* __restrict__ wo1, const float* __restrict__ wo2,
             const float* __restrict__ wo3,
             const float* __restrict__ bo1, const float* __restrict__ bo2,
             const float* __restrict__ bo3,
             float* __restrict__ out, float* aux, int t) {
  const int c  = threadIdx.x;                 // 0..63 (output feature)
  const int ty = threadIdx.y;                 // 0..3
  const int b  = blockIdx.x * 4 + ty;
  const float* h2r = h2 + (size_t)b * H_;
  const float* h3r = h3 + (size_t)b * H_;
  const float* h4r = h4 + (size_t)b * H_;
  const float* w1r = wo1 + (size_t)c * H_;
  const float* w2r = wo2 + (size_t)c * H_;
  const float* w3r = wo3 + (size_t)c * H_;
  float s1 = 0.f, s2 = 0.f, s3 = 0.f, s5 = 0.f;
#pragma unroll 4
  for (int k = 0; k < H_; ++k) {
    float w1 = w1r[k];
    s1 += h2r[k] * w1;
    s5 += h4r[k] * w1;
    s2 += h3r[k] * w2r[k];
    s3 += h4r[k] * w3r[k];
  }
  float o = tanhf(s1 + bo1[c]) + tanhf(s2 + bo2[c]) + tanhf(s3 + bo3[c]);
  out[((size_t)b * T_ + t) * OUT_ + c] = o;

  // aux3: unbiased variance over features of tanh(h4 @ w_o1^T + b_o1)
  float y = tanhf(s5 + bo1[c]);
  __shared__ float ybuf[4][64];
  ybuf[ty][c] = y;
  __syncthreads();
  if (c == 0) {
    float s = 0.f;
    for (int i = 0; i < 64; ++i) s += ybuf[ty][i];
    float m = s * (1.f / 64.f);
    float q = 0.f;
    for (int i = 0; i < 64; ++i) { float d = ybuf[ty][i] - m; q += d * d; }
    atomicAdd(aux + 2, (q * (1.f / 63.f)) * (1.f / (float)B_));
  }
}

// ---------------- utility kernels ----------------
__global__ void __launch_bounds__(256)
cvt_bf16_kernel(const float* __restrict__ s, bf16_t* __restrict__ d, int n) {
  int i = blockIdx.x * 256 + threadIdx.x;
  if (i < n) d[i] = (bf16_t)s[i];
}

__global__ void __launch_bounds__(256)
init_kernel(char* ws) {
  int i = blockIdx.x * 256 + threadIdx.x;   // 0..BH-1
#pragma unroll
  for (int l = 0; l < 4; ++l) {
    ((float*)(ws + OFF_H0 + (size_t)l * SZ_HF))[i] = 0.f;
    ((bf16_t*)(ws + OFF_HBF0 + (size_t)l * SZ_HB))[i] = (bf16_t)0.f;
  }
  if (i < 3) ((float*)(ws + OFF_AUX))[i] = 0.f;
}

__global__ void finalize_kernel(const float* aux, float* dst) {
  int i = threadIdx.x;
  if (i < 3) dst[i] = aux[i];
}

// ---------------- host launcher ----------------
extern "C" void kernel_launch(void* const* d_in, const int* in_sizes, int n_in,
                              void* d_out, int out_size, void* d_ws, size_t ws_size,
                              hipStream_t stream) {
  (void)in_sizes; (void)n_in; (void)out_size; (void)ws_size;
  char* ws = (char*)d_ws;

  const float* x = (const float*)d_in[0];
  const float *wih[4], *whh[4], *bih[4], *bhh[4];
  for (int l = 0; l < 4; ++l) {
    wih[l] = (const float*)d_in[1 + 4 * l];
    whh[l] = (const float*)d_in[2 + 4 * l];
    bih[l] = (const float*)d_in[3 + 4 * l];
    bhh[l] = (const float*)d_in[4 + 4 * l];
  }
  const float *wo[3], *bo[3];
  for (int h = 0; h < 3; ++h) {
    wo[h] = (const float*)d_in[17 + 2 * h];
    bo[h] = (const float*)d_in[18 + 2 * h];
  }
  float* out = (float*)d_out;

  bf16_t* wihb[4] = { (bf16_t*)(ws + OFF_WIHB0), (bf16_t*)(ws + OFF_WIHB1),
                      (bf16_t*)(ws + OFF_WIHB2), (bf16_t*)(ws + OFF_WIHB3) };
  bf16_t* whhb[4] = { (bf16_t*)(ws + OFF_WHHB0), (bf16_t*)(ws + OFF_WHHB1),
                      (bf16_t*)(ws + OFF_WHHB2), (bf16_t*)(ws + OFF_WHHB3) };
  bf16_t* xbf = (bf16_t*)(ws + OFF_XBF);
  float*  hF[4]; bf16_t* hB[4]; float* giF[4]; float* ghF[4];
  for (int l = 0; l < 4; ++l) {
    hF[l]  = (float*)(ws + OFF_H0   + (size_t)l * SZ_HF);
    hB[l]  = (bf16_t*)(ws + OFF_HBF0 + (size_t)l * SZ_HB);
    giF[l] = (float*)(ws + OFF_GI0  + (size_t)l * SZ_GF);
    ghF[l] = (float*)(ws + OFF_GH0  + (size_t)l * SZ_GF);
  }
  float* ring = (float*)(ws + OFF_RING);
  float* aux  = (float*)(ws + OFF_AUX);

  // one-time: weight + x conversion to bf16, state init
  cvt_bf16_kernel<<<(G3_*IN_ + 255) / 256, 256, 0, stream>>>(wih[0], wihb[0], G3_*IN_);
  for (int l = 1; l < 4; ++l)
    cvt_bf16_kernel<<<(G3_*H_ + 255) / 256, 256, 0, stream>>>(wih[l], wihb[l], G3_*H_);
  for (int l = 0; l < 4; ++l)
    cvt_bf16_kernel<<<(G3_*H_ + 255) / 256, 256, 0, stream>>>(whh[l], whhb[l], G3_*H_);
  cvt_bf16_kernel<<<(B_*T_*IN_ + 255) / 256, 256, 0, stream>>>(x, xbf, B_*T_*IN_);
  init_kernel<<<BH_ / 256, 256, 0, stream>>>(ws);

  for (int t = 0; t < T_; ++t) {
    // layer 1: gi = x_t @ w_ih1^T (K=64), gh = h1 @ w_hh1^T (K=1024)
    GemmArgs ga = {};
    ga.d[0].A = xbf + (size_t)t * IN_; ga.d[0].Bm = wihb[0]; ga.d[0].C = giF[0];
    ga.d[0].lda = T_ * IN_; ga.d[0].K = IN_;
    ga.d[1].A = hB[0]; ga.d[1].Bm = whhb[0]; ga.d[1].C = ghF[0];
    ga.d[1].lda = H_; ga.d[1].K = H_;
    wmma_gemm_kernel<<<dim3(4, 12, 2), 256, 0, stream>>>(ga, G3_);

    GateArgs g1 = {};
    g1.d[0].gi = giF[0]; g1.d[0].gh = ghF[0]; g1.d[0].bih = bih[0]; g1.d[0].bhh = bhh[0];
    g1.d[0].h = hF[0]; g1.d[0].hbf = hB[0]; g1.d[0].mode = 0;
    gru_gate_kernel<<<dim3(BH_ / 256, 1, 1), 256, 0, stream>>>(g1, ring, aux, t);

    // layers 2-4: 6 GEMMs, all K=1024, A = h1 (gi) or own h (gh)
    GemmArgs gb = {};
    for (int l = 1; l < 4; ++l) {
      GemmDesc& di = gb.d[2 * (l - 1)];
      di.A = hB[0]; di.Bm = wihb[l]; di.C = giF[l]; di.lda = H_; di.K = H_;
      GemmDesc& dh = gb.d[2 * (l - 1) + 1];
      dh.A = hB[l]; dh.Bm = whhb[l]; dh.C = ghF[l]; dh.lda = H_; dh.K = H_;
    }
    wmma_gemm_kernel<<<dim3(4, 12, 6), 256, 0, stream>>>(gb, G3_);

    GateArgs g2 = {};
    g2.d[0].gi = giF[1]; g2.d[0].gh = ghF[1]; g2.d[0].bih = bih[1]; g2.d[0].bhh = bhh[1];
    g2.d[0].h = hF[1]; g2.d[0].hbf = hB[1]; g2.d[0].mode = 1;   // aux1 lag-7
    g2.d[1].gi = giF[2]; g2.d[1].gh = ghF[2]; g2.d[1].bih = bih[2]; g2.d[1].bhh = bhh[2];
    g2.d[1].h = hF[2]; g2.d[1].hbf = hB[2]; g2.d[1].mode = 2;   // aux2 lag-1
    g2.d[2].gi = giF[3]; g2.d[2].gh = ghF[3]; g2.d[2].bih = bih[3]; g2.d[2].bhh = bhh[3];
    g2.d[2].h = hF[3]; g2.d[2].hbf = hB[3]; g2.d[2].mode = 0;
    gru_gate_kernel<<<dim3(BH_ / 256, 1, 3), 256, 0, stream>>>(g2, ring, aux, t);

    heads_kernel<<<dim3(B_ / 4, 1, 1), dim3(64, 4, 1), 0, stream>>>(
        hF[1], hF[2], hF[3], wo[0], wo[1], wo[2], bo[0], bo[1], bo[2], out, aux, t);
  }

  finalize_kernel<<<1, 4, 0, stream>>>(aux, out + (size_t)B_ * T_ * OUT_);
}
